// MixConv_14388140441689
// MI455X (gfx1250) — matrix operand order repaired and checked by reading, not compiled, not run on hardware
//
#include <hip/hip_runtime.h>
#include <hip/hip_bf16.h>
#include <math.h>

typedef __attribute__((ext_vector_type(16))) _Float16 v16h;
typedef __attribute__((ext_vector_type(8)))  _Float16 v8h;
typedef __attribute__((ext_vector_type(4)))  _Float16 v4h;
typedef __attribute__((ext_vector_type(8)))  float    v8f;
typedef __attribute__((ext_vector_type(4)))  float    v4f;

#define KNN    32
#define BN_EPS 1e-5f

#define GF_RELU   1
#define GF_ACCUM  2
#define GF_STATS  4
#define GF_GATHER 8

// ---------------------------------------------------------------------------
// Utility kernels
// ---------------------------------------------------------------------------
__global__ __launch_bounds__(256) void fill_f32(float* p, float v, int n) {
    int i = blockIdx.x * blockDim.x + threadIdx.x;
    if (i < n) p[i] = v;
}

// ---------------------------------------------------------------------------
// WMMA tile GEMM:  Out[M,Nc] (+)= op(A)[M,K] @ W[K,Nc]  (+bias, ReLU, BN stats)
// op(A): dense fp32 row-major, or gathered edge features [x_i | x_j - x_i].
// Block: 256 threads = 8 waves; block tile 64x32 (4 M-tiles x 2 N-tiles);
// wave tile 16x16, K step 32 via v_wmma_f32_16x16x32_f16.
// Assumes M % 64 == 0, Nc % 32 == 0 (all call sites satisfy this).
// ---------------------------------------------------------------------------
template <int FLAGS>
__global__ __launch_bounds__(256)
void gemm_wmma(const float* __restrict__ A,
               const float* __restrict__ xsrc,   // gather source [*, C]
               const int*   __restrict__ gidx,   // neighbor id per edge row [M]
               int C,                            // gather feature width (K = 2C)
               const float* __restrict__ W,      // [K, Nc] fp32 row-major
               const float* __restrict__ bias,   // [Nc] or nullptr
               float* __restrict__ Out,          // [M, Nc] fp32
               float* __restrict__ gsum,         // [Nc] BN sum (STATS)
               float* __restrict__ gsq,          // [Nc] BN sumsq (STATS)
               int M, int K, int Nc)
{
    constexpr bool RELU   = (FLAGS & GF_RELU)   != 0;
    constexpr bool ACC    = (FLAGS & GF_ACCUM)  != 0;
    constexpr bool STATS  = (FLAGS & GF_STATS)  != 0;
    constexpr bool GATHER = (FLAGS & GF_GATHER) != 0;

    __shared__ _Float16 As[64][40];   // A tile, row-major [row][k], 80B row stride
    __shared__ _Float16 Bs[32][40];   // B tile, TRANSPOSED [col][k]
    __shared__ float s_sum[32], s_sq[32];

    const int tid  = threadIdx.x;
    const int lane = tid & 31;
    const int wave = tid >> 5;
    const int m0b  = blockIdx.x * 64;
    const int n0b  = blockIdx.y * 32;
    const int n0   = n0b + (wave >> 2) * 16;

    if constexpr (STATS) {
        if (tid < 32) { s_sum[tid] = 0.f; s_sq[tid] = 0.f; }
    }

    // C/D layout: VGPR v, lane l -> row rbase + v, col n0 + (l&15)
    const int col   = n0 + (lane & 15);
    const int rbase = m0b + (wave & 3) * 16 + ((lane >= 16) ? 8 : 0);

    v8f acc = {};
    if constexpr (ACC) {
#pragma unroll
        for (int v = 0; v < 8; ++v)
            acc[v] = Out[(size_t)(rbase + v) * Nc + col];
    }

    for (int k0 = 0; k0 < K; k0 += 32) {
        // ---- stage A tile: 64 rows x 32 k as 512 chunks of 4 halves ----
#pragma unroll
        for (int t = tid; t < 512; t += 256) {
            const int r    = t >> 3;
            const int kg   = (t & 7) * 4;
            const int grow = m0b + r;
            const int gkb  = k0 + kg;
            v4f vv = {};
            if constexpr (!GATHER) {
                if (gkb + 3 < K) {
                    vv = *(const v4f*)&A[(size_t)grow * K + gkb];
                } else {
#pragma unroll
                    for (int u = 0; u < 4; ++u)
                        if (gkb + u < K) vv[u] = A[(size_t)grow * K + gkb + u];
                }
            } else {
                const int ni = grow >> 5;          // edge row -> source node (KNN=32)
#pragma unroll
                for (int u = 0; u < 4; ++u) {
                    const int gk = gkb + u;
                    if (gk < K) {
                        if (gk < C) {
                            vv[u] = xsrc[(size_t)ni * C + gk];
                        } else {
                            const int cc = gk - C;
                            const int nj = gidx[grow];
                            vv[u] = xsrc[(size_t)nj * C + cc] - xsrc[(size_t)ni * C + cc];
                        }
                    }
                }
            }
            v4h hv;
            hv[0] = (_Float16)vv[0]; hv[1] = (_Float16)vv[1];
            hv[2] = (_Float16)vv[2]; hv[3] = (_Float16)vv[3];
            *(v4h*)&As[r][kg] = hv;                // ds_store_b64
        }
        // ---- stage B tile transposed: 32 cols x 32 k, one 4-chunk/thread ----
        {
            const int nn  = tid >> 3;
            const int kg  = (tid & 7) * 4;
            v4h hv;
#pragma unroll
            for (int u = 0; u < 4; ++u) {
                const int gk = k0 + kg + u;
                hv[u] = (_Float16)((gk < K) ? W[(size_t)gk * Nc + n0b + nn] : 0.f);
            }
            *(v4h*)&Bs[nn][kg] = hv;               // ds_store_b64
        }
        __syncthreads();

        // ---- fragments: two aligned 16B LDS loads each (ds_load_b128) ----
        const int arow = (wave & 3) * 16 + (lane & 15);
        const int kb   = (lane < 16) ? 0 : 8;      // A: K = kb+{0..7}, kb+{16..23}
        v8h a_lo = *(const v8h*)&As[arow][kb];
        v8h a_hi = *(const v8h*)&As[arow][kb + 16];
        v16h af = __builtin_shufflevector(a_lo, a_hi,
                    0,1,2,3,4,5,6,7,8,9,10,11,12,13,14,15);

        const int brow = (wave >> 2) * 16 + (lane & 15);
        const int kb2  = (lane < 16) ? 0 : 8;      // B: K = kb2+{0..7}, kb2+{16..23}
        v8h b_lo = *(const v8h*)&Bs[brow][kb2];
        v8h b_hi = *(const v8h*)&Bs[brow][kb2 + 16];
        v16h bf = __builtin_shufflevector(b_lo, b_hi,
                    0,1,2,3,4,5,6,7,8,9,10,11,12,13,14,15);

        acc = __builtin_amdgcn_wmma_f32_16x16x32_f16(
                  false, af, false, bf, (short)0, acc, false, false);
        __syncthreads();
    }

    // ---- epilogue: bias, relu, store, BN partial stats ----
    const float b = bias ? bias[col] : 0.f;
    float lsum = 0.f, lsq = 0.f;
#pragma unroll
    for (int v = 0; v < 8; ++v) {
        float val = acc[v] + b;
        if constexpr (RELU) val = fmaxf(val, 0.f);
        Out[(size_t)(rbase + v) * Nc + col] = val;
        lsum += val; lsq = fmaf(val, val, lsq);
    }
    if constexpr (STATS) {
        atomicAdd(&s_sum[col - n0b], lsum);
        atomicAdd(&s_sq[col - n0b], lsq);
        __syncthreads();
        if (tid < 32) {
            atomicAdd(&gsum[n0b + tid], s_sum[tid]);
            atomicAdd(&gsq[n0b + tid],  s_sq[tid]);
        }
    }
}

// ---------------------------------------------------------------------------
// BatchNorm finalize (per column) and apply
// ---------------------------------------------------------------------------
__global__ __launch_bounds__(64)
void bn_finalize(const float* gsum, const float* gsq, const float* g,
                 const float* beta, float* scale, float* shift, int Nc, float invM)
{
    int c = blockIdx.x * blockDim.x + threadIdx.x;
    if (c >= Nc) return;
    float mu  = gsum[c] * invM;
    float var = gsq[c] * invM - mu * mu;
    float a   = g[c] * rsqrtf(var + BN_EPS);
    scale[c] = a;
    shift[c] = beta[c] - mu * a;
}

__global__ __launch_bounds__(256)
void bn_apply(float* h, const float* scale, const float* shift, int M, int Nc)
{
    size_t i = (size_t)blockIdx.x * blockDim.x + threadIdx.x;
    if (i >= (size_t)M * Nc) return;
    int c = (int)(i % Nc);
    h[i] = fmaf(h[i], scale[c], shift[c]);
}

// ---------------------------------------------------------------------------
// Max over k neighbors: out[n,c] = max_k h[n*KNN+k, c]
// ---------------------------------------------------------------------------
__global__ __launch_bounds__(256)
void max_over_k(const float* __restrict__ h, float* __restrict__ out, int Nn, int Nc)
{
    int i = blockIdx.x * blockDim.x + threadIdx.x;
    if (i >= Nn * Nc) return;
    int n = i / Nc, c = i % Nc;
    const float* p = h + (size_t)n * KNN * Nc + c;
    float m = -__builtin_inff();
#pragma unroll 4
    for (int k = 0; k < KNN; ++k) m = fmaxf(m, p[(size_t)k * Nc]);
    out[i] = m;
}

// ---------------------------------------------------------------------------
// Brute-force top-32 kNN (self excluded), D = 3 or 64, LDS-staged candidates
// ---------------------------------------------------------------------------
template <int D>
__global__ __launch_bounds__(128)
void knn_kernel(const float* __restrict__ X, int* __restrict__ outIdx, int Nn)
{
    __shared__ float tile[128 * D];
    int q = blockIdx.x * 128 + threadIdx.x;
    if (q >= Nn) q = Nn - 1;                 // uniform work, no barrier divergence

    float qv[D];
#pragma unroll
    for (int d = 0; d < D; ++d) qv[d] = X[(size_t)q * D + d];

    float dist[KNN]; int idc[KNN];
#pragma unroll
    for (int t = 0; t < KNN; ++t) { dist[t] = __builtin_inff(); idc[t] = -1; }
    float worst = __builtin_inff(); int wslot = 0;

    for (int j0 = 0; j0 < Nn; j0 += 128) {
        __syncthreads();
        for (int t = threadIdx.x; t < 128 * D; t += 128)
            tile[t] = X[(size_t)j0 * D + t];
        __syncthreads();
        for (int jj = 0; jj < 128; ++jj) {
            int j = j0 + jj;
            float d2 = 0.f;
#pragma unroll
            for (int d = 0; d < D; ++d) {
                float df = qv[d] - tile[jj * D + d];
                d2 = fmaf(df, df, d2);
            }
            if (j == q) continue;
            if (d2 < worst) {
                dist[wslot] = d2; idc[wslot] = j;
                worst = dist[0]; wslot = 0;
#pragma unroll
                for (int t = 1; t < KNN; ++t)
                    if (dist[t] > worst) { worst = dist[t]; wslot = t; }
            }
        }
    }
    int* op = outIdx + (size_t)q * KNN;
#pragma unroll
    for (int t = 0; t < KNN; ++t) op[t] = idc[t];
}

// ---------------------------------------------------------------------------
// TAGConv support
// ---------------------------------------------------------------------------
__global__ __launch_bounds__(256)
void deg_kernel(const int* __restrict__ col, float* deg, int E)
{
    int e = blockIdx.x * blockDim.x + threadIdx.x;
    if (e < E) atomicAdd(&deg[col[e]], 1.0f);
}

__global__ __launch_bounds__(256)
void dinv_kernel(float* deg, int Nn)
{
    int i = blockIdx.x * blockDim.x + threadIdx.x;
    if (i < Nn) {
        float d = deg[i];
        deg[i] = (d > 0.f) ? rsqrtf(fmaxf(d, 1.f)) : 0.f;
    }
}

__global__ __launch_bounds__(256)
void edge_norm_kernel(const int* __restrict__ row, const int* __restrict__ col,
                      const float* __restrict__ dinv, float* enorm, int E)
{
    int e = blockIdx.x * blockDim.x + threadIdx.x;
    if (e < E) enorm[e] = dinv[row[e]] * dinv[col[e]];
}

__global__ __launch_bounds__(256)
void scatter_kernel(const float* __restrict__ src, float* __restrict__ dst,
                    const int* __restrict__ row, const int* __restrict__ col,
                    const float* __restrict__ enorm, int E, int C)
{
    int i = blockIdx.x * blockDim.x + threadIdx.x;
    if (i >= E * C) return;
    int e = i / C, c = i % C;
    atomicAdd(&dst[(size_t)col[e] * C + c], enorm[e] * src[(size_t)row[e] * C + c]);
}

__global__ __launch_bounds__(256)
void bias_relu_kernel(float* h, const float* bias, int M, int Nc)
{
    int i = blockIdx.x * blockDim.x + threadIdx.x;
    if (i >= M * Nc) return;
    int c = i % Nc;
    h[i] = fmaxf(h[i] + bias[c], 0.f);
}

__global__ __launch_bounds__(256)
void concat2(const float* __restrict__ a, int Ca, const float* __restrict__ b,
             int Cb, float* __restrict__ dst, int Nn)
{
    int Cd = Ca + Cb;
    int i = blockIdx.x * blockDim.x + threadIdx.x;
    if (i >= Nn * Cd) return;
    int n = i / Cd, c = i % Cd;
    dst[i] = (c < Ca) ? a[(size_t)n * Ca + c] : b[(size_t)n * Cb + (c - Ca)];
}

__global__ __launch_bounds__(256)
void final_linear(const float* __restrict__ h, const float* __restrict__ W,
                  const float* __restrict__ b, float* __restrict__ out, int Nn, int C)
{
    int n = blockIdx.x * blockDim.x + threadIdx.x;
    if (n >= Nn) return;
    float s = b[0];
    for (int c = 0; c < C; ++c) s = fmaf(h[(size_t)n * C + c], W[c], s);
    out[n] = s;
}

// ---------------------------------------------------------------------------
// Host orchestration
// ---------------------------------------------------------------------------
extern "C" void kernel_launch(void* const* d_in, const int* in_sizes, int n_in,
                              void* d_out, int out_size, void* d_ws, size_t ws_size,
                              hipStream_t stream)
{
    const int Nn = in_sizes[0] / 3;   // 8192
    const int E  = in_sizes[2] / 2;   // 131072
    const int ME = Nn * KNN;          // 262144

    const float* pos       = (const float*)d_in[0];
    const float* xin       = (const float*)d_in[1];
    const int*   ei        = (const int*)d_in[2];
    const float* mlp1_W1   = (const float*)d_in[3];
    const float* mlp1_W23  = (const float*)d_in[4];
    const float* mlp1_b    = (const float*)d_in[5];
    const float* mlp1_g    = (const float*)d_in[6];
    const float* mlp1_beta = (const float*)d_in[7];
    const float* mlp2_W    = (const float*)d_in[8];
    const float* mlp2_b    = (const float*)d_in[9];
    const float* mlp2_g    = (const float*)d_in[10];
    const float* mlp2_beta = (const float*)d_in[11];
    const float* lin1_W    = (const float*)d_in[12];
    const float* lin1_b    = (const float*)d_in[13];
    const float* lin1_g    = (const float*)d_in[14];
    const float* lin1_beta = (const float*)d_in[15];
    const float* tag1_W    = (const float*)d_in[16];
    const float* tag1_b    = (const float*)d_in[17];
    const float* tag2_W    = (const float*)d_in[18];
    const float* tag2_b    = (const float*)d_in[19];
    const float* lin2_W    = (const float*)d_in[20];
    const float* lin2_b    = (const float*)d_in[21];
    const float* lin2_g    = (const float*)d_in[22];
    const float* lin2_beta = (const float*)d_in[23];
    const float* mix1_W    = (const float*)d_in[24];
    const float* mix1_b    = (const float*)d_in[25];
    const float* mix1_g    = (const float*)d_in[26];
    const float* mix1_beta = (const float*)d_in[27];
    const float* mix2_W    = (const float*)d_in[28];
    const float* mix2_b    = (const float*)d_in[29];
    const float* mix2_g    = (const float*)d_in[30];
    const float* mix2_beta = (const float*)d_in[31];
    const float* out_W     = (const float*)d_in[32];
    const float* out_b     = (const float*)d_in[33];
    const int* row = ei;
    const int* col = ei + E;
    float* y = (float*)d_out;

    // --- workspace bump allocator ---
    char* wsp = (char*)d_ws;
    auto balloc = [&](size_t bytes) -> char* {
        char* p = wsp; wsp += (bytes + 255) & ~(size_t)255; return p;
    };
    float* BIG     = (float*)balloc((size_t)ME * 128 * sizeof(float)); // 134 MB
    int*   idx1    = (int*)  balloc((size_t)ME * sizeof(int));
    int*   idx2    = (int*)  balloc((size_t)ME * sizeof(int));
    float* x1      = (float*)balloc((size_t)Nn * 64 * sizeof(float));
    float* x2      = (float*)balloc((size_t)Nn * 128 * sizeof(float));
    float* cat192a = (float*)balloc((size_t)Nn * 192 * sizeof(float));
    float* out_dg  = (float*)balloc((size_t)Nn * 512 * sizeof(float));
    float* deg     = (float*)balloc((size_t)Nn * sizeof(float));
    float* enorm   = (float*)balloc((size_t)E * sizeof(float));
    float* ssum    = (float*)balloc(512 * sizeof(float));
    float* ssq     = (float*)balloc(512 * sizeof(float));
    float* sscale  = (float*)balloc(512 * sizeof(float));
    float* sshift  = (float*)balloc(512 * sizeof(float));

    // overlays inside BIG (DGCNN edge activations dead after x2 extracted)
    float* hA      = BIG;                       // ME*64
    float* hB      = BIG + (size_t)ME * 64;     // ME*64
    float* hBIG    = BIG;                       // ME*128
    float* th1     = BIG;                       // N*128
    float* th2     = th1 + (size_t)Nn * 128;
    float* g1      = th2 + (size_t)Nn * 128;    // N*64
    float* g2      = g1  + (size_t)Nn * 64;     // N*128
    float* cat192b = g2  + (size_t)Nn * 128;    // N*192
    float* out_g   = cat192b + (size_t)Nn * 192;// N*512
    float* cat1024 = out_g   + (size_t)Nn * 512;// N*1024
    float* h512    = cat1024 + (size_t)Nn * 1024;
    float* h256    = h512    + (size_t)Nn * 512;

    auto zero = [&](float* p, size_t n) {
        fill_f32<<<dim3((unsigned)((n + 255) / 256)), 256, 0, stream>>>(p, 0.f, (int)n);
    };
    auto gemm = [&](const float* A, const float* xs, const int* gi, int C,
                    const float* W, const float* bias, float* Out,
                    int M, int K, int Nc, int flags) {
        dim3 g((unsigned)(M / 64), (unsigned)(Nc / 32));
        switch (flags) {
        case GF_RELU | GF_STATS | GF_GATHER:
            gemm_wmma<GF_RELU | GF_STATS | GF_GATHER><<<g, 256, 0, stream>>>(
                A, xs, gi, C, W, bias, Out, ssum, ssq, M, K, Nc);
            break;
        case GF_RELU | GF_STATS:
            gemm_wmma<GF_RELU | GF_STATS><<<g, 256, 0, stream>>>(
                A, xs, gi, C, W, bias, Out, ssum, ssq, M, K, Nc);
            break;
        case GF_ACCUM:
            gemm_wmma<GF_ACCUM><<<g, 256, 0, stream>>>(
                A, xs, gi, C, W, bias, Out, ssum, ssq, M, K, Nc);
            break;
        default:
            gemm_wmma<0><<<g, 256, 0, stream>>>(
                A, xs, gi, C, W, bias, Out, ssum, ssq, M, K, Nc);
            break;
        }
    };
    auto bn = [&](float* h, int M, int Nc, const float* g, const float* beta) {
        bn_finalize<<<dim3((unsigned)((Nc + 63) / 64)), 64, 0, stream>>>(
            ssum, ssq, g, beta, sscale, sshift, Nc, 1.0f / (float)M);
        size_t tot = (size_t)M * Nc;
        bn_apply<<<dim3((unsigned)((tot + 255) / 256)), 256, 0, stream>>>(
            h, sscale, sshift, M, Nc);
    };

    // ================= DGCNN branch =================
    knn_kernel<3><<<dim3((unsigned)(Nn / 128)), 128, 0, stream>>>(pos, idx1, Nn);

    // edge MLP1: 6 -> 64 -> 64 -> 64 (gathered A, fused BN stats)
    zero(ssum, 64); zero(ssq, 64);
    gemm(nullptr, pos, idx1, 3, mlp1_W1, mlp1_b, hA, ME, 6, 64,
         GF_RELU | GF_STATS | GF_GATHER);
    bn(hA, ME, 64, mlp1_g, mlp1_beta);

    zero(ssum, 64); zero(ssq, 64);
    gemm(hA, nullptr, nullptr, 0, mlp1_W23, mlp1_b + 64, hB, ME, 64, 64,
         GF_RELU | GF_STATS);
    bn(hB, ME, 64, mlp1_g + 64, mlp1_beta + 64);

    zero(ssum, 64); zero(ssq, 64);
    gemm(hB, nullptr, nullptr, 0, mlp1_W23 + 64 * 64, mlp1_b + 128, hA, ME, 64, 64,
         GF_RELU | GF_STATS);
    bn(hA, ME, 64, mlp1_g + 128, mlp1_beta + 128);

    max_over_k<<<dim3((unsigned)((Nn * 64 + 255) / 256)), 256, 0, stream>>>(hA, x1, Nn, 64);

    knn_kernel<64><<<dim3((unsigned)(Nn / 128)), 128, 0, stream>>>(x1, idx2, Nn);

    // edge MLP2: 128 -> 128
    zero(ssum, 128); zero(ssq, 128);
    gemm(nullptr, x1, idx2, 64, mlp2_W, mlp2_b, hBIG, ME, 128, 128,
         GF_RELU | GF_STATS | GF_GATHER);
    bn(hBIG, ME, 128, mlp2_g, mlp2_beta);

    max_over_k<<<dim3((unsigned)((Nn * 128 + 255) / 256)), 256, 0, stream>>>(hBIG, x2, Nn, 128);

    // lin1: [x1|x2] (192) -> 512
    concat2<<<dim3((unsigned)((Nn * 192 + 255) / 256)), 256, 0, stream>>>(x1, 64, x2, 128, cat192a, Nn);
    zero(ssum, 512); zero(ssq, 512);
    gemm(cat192a, nullptr, nullptr, 0, lin1_W, lin1_b, out_dg, Nn, 192, 512,
         GF_RELU | GF_STATS);
    bn(out_dg, Nn, 512, lin1_g, lin1_beta);

    // ================= TAGConv branch =================
    zero(deg, (size_t)Nn);
    deg_kernel<<<dim3((unsigned)((E + 255) / 256)), 256, 0, stream>>>(col, deg, E);
    dinv_kernel<<<dim3((unsigned)((Nn + 255) / 256)), 256, 0, stream>>>(deg, Nn);
    edge_norm_kernel<<<dim3((unsigned)((E + 255) / 256)), 256, 0, stream>>>(row, col, deg, enorm, E);

    // tag1: x(4) -> 64, K=3 hops
    gemm(xin, nullptr, nullptr, 0, tag1_W, nullptr, g1, Nn, 4, 64, 0);
    zero(th2, (size_t)Nn * 4);
    scatter_kernel<<<dim3((unsigned)((E * 4 + 255) / 256)), 256, 0, stream>>>(xin, th2, row, col, enorm, E, 4);
    gemm(th2, nullptr, nullptr, 0, tag1_W + 1 * 4 * 64, nullptr, g1, Nn, 4, 64, GF_ACCUM);
    zero(th1, (size_t)Nn * 4);
    scatter_kernel<<<dim3((unsigned)((E * 4 + 255) / 256)), 256, 0, stream>>>(th2, th1, row, col, enorm, E, 4);
    gemm(th1, nullptr, nullptr, 0, tag1_W + 2 * 4 * 64, nullptr, g1, Nn, 4, 64, GF_ACCUM);
    zero(th2, (size_t)Nn * 4);
    scatter_kernel<<<dim3((unsigned)((E * 4 + 255) / 256)), 256, 0, stream>>>(th1, th2, row, col, enorm, E, 4);
    gemm(th2, nullptr, nullptr, 0, tag1_W + 3 * 4 * 64, nullptr, g1, Nn, 4, 64, GF_ACCUM);
    bias_relu_kernel<<<dim3((unsigned)((Nn * 64 + 255) / 256)), 256, 0, stream>>>(g1, tag1_b, Nn, 64);

    // tag2: g1(64) -> 128, K=3 hops
    gemm(g1, nullptr, nullptr, 0, tag2_W, nullptr, g2, Nn, 64, 128, 0);
    zero(th2, (size_t)Nn * 64);
    scatter_kernel<<<dim3((unsigned)((E * 64 + 255) / 256)), 256, 0, stream>>>(g1, th2, row, col, enorm, E, 64);
    gemm(th2, nullptr, nullptr, 0, tag2_W + 1 * 64 * 128, nullptr, g2, Nn, 64, 128, GF_ACCUM);
    zero(th1, (size_t)Nn * 64);
    scatter_kernel<<<dim3((unsigned)((E * 64 + 255) / 256)), 256, 0, stream>>>(th2, th1, row, col, enorm, E, 64);
    gemm(th1, nullptr, nullptr, 0, tag2_W + 2 * 64 * 128, nullptr, g2, Nn, 64, 128, GF_ACCUM);
    zero(th2, (size_t)Nn * 64);
    scatter_kernel<<<dim3((unsigned)((E * 64 + 255) / 256)), 256, 0, stream>>>(th1, th2, row, col, enorm, E, 64);
    gemm(th2, nullptr, nullptr, 0, tag2_W + 3 * 64 * 128, nullptr, g2, Nn, 64, 128, GF_ACCUM);
    bias_relu_kernel<<<dim3((unsigned)((Nn * 128 + 255) / 256)), 256, 0, stream>>>(g2, tag2_b, Nn, 128);

    // lin2: [g1|g2] (192) -> 512
    concat2<<<dim3((unsigned)((Nn * 192 + 255) / 256)), 256, 0, stream>>>(g1, 64, g2, 128, cat192b, Nn);
    zero(ssum, 512); zero(ssq, 512);
    gemm(cat192b, nullptr, nullptr, 0, lin2_W, lin2_b, out_g, Nn, 192, 512,
         GF_RELU | GF_STATS);
    bn(out_g, Nn, 512, lin2_g, lin2_beta);

    // ================= mix head =================
    concat2<<<dim3((unsigned)((Nn * 1024 + 255) / 256)), 256, 0, stream>>>(out_dg, 512, out_g, 512, cat1024, Nn);
    zero(ssum, 512); zero(ssq, 512);
    gemm(cat1024, nullptr, nullptr, 0, mix1_W, mix1_b, h512, Nn, 1024, 512,
         GF_RELU | GF_STATS);
    bn(h512, Nn, 512, mix1_g, mix1_beta);

    zero(ssum, 256); zero(ssq, 256);
    gemm(h512, nullptr, nullptr, 0, mix2_W, mix2_b, h256, Nn, 512, 256,
         GF_RELU | GF_STATS);
    bn(h256, Nn, 256, mix2_g, mix2_beta);

    final_linear<<<dim3((unsigned)((Nn + 255) / 256)), 256, 0, stream>>>(h256, out_W, out_b, y, Nn, 256);
}
